// HyenaOperator_55551107006713
// MI455X (gfx1250) — matrix operand order repaired and checked by reading, not compiled, not run on hardware
//
#include <hip/hip_runtime.h>
#include <hip/hip_bf16.h>

typedef __attribute__((ext_vector_type(16))) _Float16 v16h;
typedef __attribute__((ext_vector_type(8)))  _Float16 v8h;
typedef __attribute__((ext_vector_type(8)))  float    v8f;

constexpr int Bc = 4;
constexpr int Lc = 4096;
constexpr int Dc = 768;
constexpr int C3 = 3 * Dc;     // 2304
constexpr int EMBc = 33;
constexpr int FFNc = 64;
constexpr int KF  = 256;       // truncated FIR taps (transient underflows to 0 in fp32)
constexpr float FREQc = 10.0f;

constexpr int BM = 128, BN = 64, BK = 32;
constexpr int LDA = 40;        // padded halfs per LDS tile row (80 B, 16B-aligned)

#define USE_TDM 0
#if defined(__has_builtin)
#if __has_builtin(__builtin_amdgcn_tensor_load_to_lds) && __has_builtin(__builtin_amdgcn_s_wait_tensorcnt)
#undef USE_TDM
#define USE_TDM 1
#endif
#endif

#if USE_TDM
typedef unsigned int __attribute__((ext_vector_type(4))) u32x4;
typedef int          __attribute__((ext_vector_type(4))) i32x4;
typedef int          __attribute__((ext_vector_type(8))) i32x8;

// TDM 2D tile load: tile_rows x 32 f16 elements, row stride = row_stride_elems,
// destination LDS rows padded to 80B (pad_interval=16 DW, pad_amount=4 DW).
// D# packing per CDNA5 ISA.
__device__ static inline void tdm_load_tile_f16(
    unsigned lds_addr, const _Float16* gsrc,
    int tile_rows, int tile_cols, long row_stride_elems)
{
    unsigned long long ga = (unsigned long long)gsrc;
    u32x4 g0;
    g0[0] = 1u;                                     // count=1 (valid user D#)
    g0[1] = lds_addr;                               // lds_addr [63:32]
    g0[2] = (unsigned)(ga & 0xffffffffull);         // global_addr [95:64]
    g0[3] = (unsigned)((ga >> 32) & 0x01ffffffull)  // global_addr [120:96]
          | (2u << 30);                             // type=2 ("image")
    unsigned td0 = (unsigned)row_stride_elems;      // tensor_dim0 (row length)
    unsigned td1 = 1u << 20;                        // tensor_dim1 (ample rows)
    unsigned long long s0 = (unsigned long long)row_stride_elems;
    i32x8 g1;
    g1[0] = (1 << 16)       // data_size = 1 -> 2 bytes
          | (1 << 20)       // pad_enable
          | (3 << 22)       // pad_interval code 3 -> every 16 DWORDs (=64B row)
          | (3 << 25);      // pad_amount   code 3 -> 4 DWORDs (=16B)
    g1[1] = (int)((td0 & 0xffffu) << 16);                       // dim0[15:0] @63:48
    g1[2] = (int)((td0 >> 16) | ((td1 & 0xffffu) << 16));       // dim0[31:16], dim1[15:0]
    g1[3] = (int)((td1 >> 16) | ((unsigned)tile_cols << 16));   // dim1[31:16], tile_dim0
    g1[4] = tile_rows & 0xffff;                                 // tile_dim1 (tile_dim2=0)
    g1[5] = (int)(s0 & 0xffffffffull);                          // dim0_stride[31:0]
    g1[6] = (int)((s0 >> 32) & 0xffffull);                      // dim0_stride[47:32]
    g1[7] = 0;
    i32x4 z4 = {0, 0, 0, 0};
#if __clang_major__ >= 23
    i32x8 z8 = {0, 0, 0, 0, 0, 0, 0, 0};
    __builtin_amdgcn_tensor_load_to_lds(g0, g1, z4, z4, z8, 0);
#else
    __builtin_amdgcn_tensor_load_to_lds(g0, g1, z4, z4, 0);
#endif
}
#endif // USE_TDM

// ---------------------------------------------------------------------------
// Elementwise f32 -> f16 convert
// ---------------------------------------------------------------------------
__global__ __launch_bounds__(256) void convert_f16(
    const float* __restrict__ src, _Float16* __restrict__ dst, long n)
{
    long i = (long)blockIdx.x * 256 + threadIdx.x;
    if (i < n) dst[i] = (_Float16)src[i];
}

// ---------------------------------------------------------------------------
// Tiled transpose + convert: src (R x C) f32 -> dst (C x R) f16
// ---------------------------------------------------------------------------
__global__ __launch_bounds__(256) void transpose_convert_f16(
    const float* __restrict__ src, long src_batch,
    _Float16* __restrict__ dst, long dst_batch, int R, int C)
{
    int bz = blockIdx.z;
    src += (long)bz * src_batch;
    dst += (long)bz * dst_batch;
    __shared__ float t[32][33];
    int c0 = blockIdx.x * 32, r0 = blockIdx.y * 32;
    int tid = threadIdx.x;
    for (int i = tid; i < 1024; i += 256) {
        int rr = i >> 5, cc = i & 31;
        t[rr][cc] = src[(long)(r0 + rr) * C + c0 + cc];
    }
    __syncthreads();
    for (int i = tid; i < 1024; i += 256) {
        int cc = i >> 5, rr = i & 31;
        dst[(long)(c0 + cc) * R + r0 + rr] = (_Float16)t[rr][cc];
    }
}

// ---------------------------------------------------------------------------
// Filter construction: g[c][k] = mlp(k)[c] * exp(-exp(f_decay[c]) * k*L/(L-1))
// Only k < KF rows needed (the decay underflows to exactly 0 in fp32 after).
// ---------------------------------------------------------------------------
__global__ __launch_bounds__(256) void build_filter(
    const float* __restrict__ W1, const float* __restrict__ b1,
    const float* __restrict__ W2, const float* __restrict__ b2,
    const float* __restrict__ f_decay, float* __restrict__ g)
{
    int l = blockIdx.x;
    int tid = threadIdx.x;
    __shared__ float pe[EMBc];
    __shared__ float h1[FFNc];
    float t = (float)l / (float)(Lc - 1);
    if (tid < EMBc) pe[tid] = sinf(t * (float)tid * FREQc);
    __syncthreads();
    if (tid < FFNc) {
        float s = b1[tid];
        for (int j = 0; j < EMBc; ++j) s += pe[j] * W1[j * FFNc + tid];
        h1[tid] = sinf(FREQc * s);
    }
    __syncthreads();
    float tw = (float)l * ((float)Lc / (float)(Lc - 1));
    for (int c = tid; c < 2 * Dc; c += 256) {
        float s = b2[c];
        #pragma unroll 8
        for (int f = 0; f < FFNc; ++f) s += h1[f] * W2[f * (2 * Dc) + c];
        float decay = expf(-expf(f_decay[c]) * tw);
        g[(long)c * KF + l] = s * decay;
    }
}

// ---------------------------------------------------------------------------
// WMMA GEMM, all-f16 operands: C = A(MxK) * BT(NxK)^T + bias
// Double-buffered TDM staging: the k+1 tile pair is issued before waiting on
// the k pair (s_wait_tensorcnt 2 passes with the 2 newest ops in flight, per
// the in-order TENSORcnt semantics), so the DMA overlaps the WMMA math.
// Fragments load as two aligned 16B ds reads each (ISA 7.12.2 layouts).
// Block = 8 waves, block tile 128x64, wave tile 32x32 (2x2 v_wmma).
// ---------------------------------------------------------------------------
__global__ __launch_bounds__(256) void gemm_wmma_f16(
    const _Float16* __restrict__ A, long a_batch,
    const _Float16* __restrict__ BT,
    const float* __restrict__ bias,
    float* __restrict__ C, long c_batch,
    int M, int N, int K)
{
    int bz = blockIdx.z;
    A += (long)bz * a_batch;
    C += (long)bz * c_batch;
    int m0 = blockIdx.y * BM;
    int n0 = blockIdx.x * BN;

    __shared__ _Float16 sA[2][BM * LDA];   // 2 x 10240 B
    __shared__ _Float16 sB[2][BN * LDA];   // 2 x  5120 B

    int tid   = threadIdx.x;
    int lane  = tid & 31;
    int wave  = tid >> 5;
    int wm    = (wave >> 1) * 32;       // 0,32,64,96
    int wn    = (wave & 1) * 32;        // 0,32
    int khalf = lane >> 4;
    int l16   = lane & 15;

    const _Float16* Ab = A + (long)m0 * K;
    const _Float16* Bb = BT + (long)n0 * K;

#if USE_TDM
    unsigned ldsA_addr[2] = {
        (unsigned)(unsigned long long)(void*)sA[0],
        (unsigned)(unsigned long long)(void*)sA[1] };
    unsigned ldsB_addr[2] = {
        (unsigned)(unsigned long long)(void*)sB[0],
        (unsigned)(unsigned long long)(void*)sB[1] };
    // Prologue: fill buffer 0
    if (tid == 0) {
        tdm_load_tile_f16(ldsA_addr[0], Ab, BM, BK, K);
        tdm_load_tile_f16(ldsB_addr[0], Bb, BN, BK, K);
    }
#endif

    v8f acc[2][2] = {};
    int cur = 0;

    for (int k0 = 0; k0 < K; k0 += BK) {
#if USE_TDM
        if (tid == 0) {
            if (k0 + BK < K) {
                // Issue next stage into the other buffer, then wait only for
                // the *older* pair: 2 newest tensor ops may remain in flight.
                tdm_load_tile_f16(ldsA_addr[cur ^ 1], Ab + (k0 + BK), BM, BK, K);
                tdm_load_tile_f16(ldsB_addr[cur ^ 1], Bb + (k0 + BK), BN, BK, K);
                __builtin_amdgcn_s_wait_tensorcnt(2);
            } else {
                __builtin_amdgcn_s_wait_tensorcnt(0);
            }
        }
#else
        for (int i = tid; i < BM * BK; i += 256) {
            int r = i >> 5, c = i & 31;
            sA[cur][r * LDA + c] = Ab[(long)r * K + k0 + c];
        }
        for (int i = tid; i < BN * BK; i += 256) {
            int r = i >> 5, c = i & 31;
            sB[cur][r * LDA + c] = Bb[(long)r * K + k0 + c];
        }
#endif
        __syncthreads();

        // Fragments: contiguous 16B chunks thanks to padded stride-40 rows.
        // A elem e -> k = e + 8*khalf + (e&8 ? 8 : 0): chunks [8h..] and [16+8h..]
        // B elem e -> k = e + 8*khalf + (e&8 ? 8 : 0): chunks [8h..] and [16+8h..]
        const _Float16* sAc = sA[cur];
        const _Float16* sBc = sB[cur];
        v16h afrag[2], bfrag[2];
        #pragma unroll
        for (int tfr = 0; tfr < 2; ++tfr) {
            const _Float16* arow = sAc + (wm + tfr * 16 + l16) * LDA;
            v8h alo = *(const v8h*)(arow + 8 * khalf);
            v8h ahi = *(const v8h*)(arow + 16 + 8 * khalf);
            afrag[tfr] = __builtin_shufflevector(alo, ahi,
                0,1,2,3,4,5,6,7,8,9,10,11,12,13,14,15);
            const _Float16* brow = sBc + (wn + tfr * 16 + l16) * LDA;
            v8h blo = *(const v8h*)(brow + 8 * khalf);
            v8h bhi = *(const v8h*)(brow + 16 + 8 * khalf);
            bfrag[tfr] = __builtin_shufflevector(blo, bhi,
                0,1,2,3,4,5,6,7,8,9,10,11,12,13,14,15);
        }
        #pragma unroll
        for (int i = 0; i < 2; ++i)
            #pragma unroll
            for (int j = 0; j < 2; ++j)
                acc[i][j] = __builtin_amdgcn_wmma_f32_16x16x32_f16(
                    false, afrag[i], false, bfrag[j],
                    (short)0, acc[i][j], false, false);
        __syncthreads();   // all waves done reading buf[cur] before it is
                           // re-targeted by the DMA two stages later
        cur ^= 1;
    }

    // C/D layout: VGPR r -> row = r + 8*khalf, col = lane&15
    #pragma unroll
    for (int i = 0; i < 2; ++i)
        #pragma unroll
        for (int j = 0; j < 2; ++j) {
            int col = n0 + wn + j * 16 + l16;
            float bv = bias[col];
            #pragma unroll
            for (int r = 0; r < 8; ++r) {
                int row = m0 + wm + i * 16 + r + 8 * khalf;
                C[(long)row * N + col] = acc[i][j][r] + bv;
            }
        }
}

// ---------------------------------------------------------------------------
// Depthwise causal conv3 along L + transpose (B,L,3D) -> (B,3,D,L).
// ---------------------------------------------------------------------------
__global__ __launch_bounds__(256) void conv3_transpose(
    const float* __restrict__ proj, const float* __restrict__ ck,
    const float* __restrict__ cb, float* __restrict__ xv)
{
    int b  = blockIdx.z;
    int c0 = blockIdx.x * 32;
    int l0 = blockIdx.y * 32;
    __shared__ float tile[34][33];
    int tid = threadIdx.x;
    for (int i = tid; i < 34 * 32; i += 256) {
        int r = i >> 5, c = i & 31;
        int l = l0 - 2 + r;
        tile[r][c] = (l >= 0) ? proj[((long)b * Lc + l) * C3 + c0 + c] : 0.0f;
    }
    __syncthreads();
    int ll = tid & 31;
    for (int cc = tid >> 5; cc < 32; cc += 8) {
        int c = c0 + cc;
        float r = tile[ll][cc]     * ck[c]
                + tile[ll + 1][cc] * ck[C3 + c]
                + tile[ll + 2][cc] * ck[2 * C3 + c] + cb[c];
        int o = c / Dc, d = c % Dc;
        xv[(((long)b * 3 + o) * Dc + d) * Lc + l0 + ll] = r;
    }
}

// ---------------------------------------------------------------------------
// Inclusive prefix scan along L, one block per (b,d) row.
// ---------------------------------------------------------------------------
__global__ __launch_bounds__(256) void row_scan(
    const float* __restrict__ v, long v_bstride, float* __restrict__ S)
{
    int rowid = blockIdx.x;
    int b = rowid / Dc, d = rowid % Dc;
    const float* vr = v + (long)b * v_bstride + (long)d * Lc;
    float* sr = S + (long)rowid * Lc;
    __shared__ float tmp[256];
    int tid = threadIdx.x;
    float run = 0.0f;
    for (int c0 = 0; c0 < Lc; c0 += 256) {
        tmp[tid] = vr[c0 + tid];
        __syncthreads();
        for (int off = 1; off < 256; off <<= 1) {
            float t = (tid >= off) ? tmp[tid - off] : 0.0f;
            __syncthreads();
            tmp[tid] += t;
            __syncthreads();
        }
        sr[c0 + tid] = tmp[tid] + run;
        float tot = tmp[255];
        __syncthreads();
        run += tot;
    }
}

// ---------------------------------------------------------------------------
// y[l] = f_bias*S[l] + sum_{k<KF} g[k]*v[l-k];  out[l] = y[l] * x[l]
// ---------------------------------------------------------------------------
__global__ __launch_bounds__(256) void fir_combine(
    const float* __restrict__ vin, long vin_bs,
    const float* __restrict__ S,
    const float* __restrict__ x, long x_bs,
    const float* __restrict__ g,
    const float* __restrict__ fb,
    float* __restrict__ out, long out_bs)
{
    int b  = blockIdx.z;
    int d  = blockIdx.y;
    int l0 = blockIdx.x * 256;
    int tid = threadIdx.x;
    const float* vr = vin + (long)b * vin_bs + (long)d * Lc;
    __shared__ float vwin[512];
    __shared__ float gk[KF];
    gk[tid] = g[(long)d * KF + tid];
    int lp = l0 - 256 + tid;
    vwin[tid]       = (lp >= 0) ? vr[lp] : 0.0f;
    vwin[256 + tid] = vr[l0 + tid];
    __syncthreads();
    float acc = 0.0f;
    #pragma unroll 8
    for (int k = 0; k < KF; ++k)
        acc = fmaf(gk[k], vwin[256 + tid - k], acc);
    long li = (long)l0 + tid;
    float y = fmaf(fb[d], S[((long)b * Dc + d) * Lc + li], acc);
    out[(long)b * out_bs + (long)d * Lc + li] =
        y * x[(long)b * x_bs + (long)d * Lc + li];
}

// ---------------------------------------------------------------------------
extern "C" void kernel_launch(void* const* d_in, const int* in_sizes, int n_in,
                              void* d_out, int out_size, void* d_ws, size_t ws_size,
                              hipStream_t stream)
{
    (void)in_sizes; (void)n_in; (void)out_size; (void)ws_size;
    const float* u      = (const float*)d_in[0];
    const float* W_in   = (const float*)d_in[1];
    const float* b_in   = (const float*)d_in[2];
    const float* conv_k = (const float*)d_in[3];
    const float* conv_b = (const float*)d_in[4];
    const float* W1     = (const float*)d_in[5];
    const float* b1     = (const float*)d_in[6];
    const float* W2     = (const float*)d_in[7];
    const float* b2     = (const float*)d_in[8];
    const float* f_bias = (const float*)d_in[9];
    const float* f_decay= (const float*)d_in[10];
    const float* W_out  = (const float*)d_in[11];
    const float* b_out  = (const float*)d_in[12];
    float* out = (float*)d_out;

    char* ws = (char*)d_ws;
    size_t off = 0;
    auto alloc = [&](size_t bytes) {
        void* p = ws + off;
        off += (bytes + 255) & ~(size_t)255;
        return p;
    };
    const long DL  = (long)Dc * Lc;
    const long XV3 = 3 * DL;
    const long BLD = (long)Bc * Lc * Dc;

    float*    proj  = (float*)   alloc(sizeof(float)    * (size_t)Bc * Lc * C3);
    float*    xv    = (float*)   alloc(sizeof(float)    * (size_t)Bc * 3 * Dc * Lc);
    float*    Sb    = (float*)   alloc(sizeof(float)    * (size_t)Bc * Dc * Lc);
    float*    yb    = (float*)   alloc(sizeof(float)    * (size_t)Bc * Dc * Lc);
    float*    g     = (float*)   alloc(sizeof(float)    * (size_t)2 * Dc * KF);
    _Float16* uh    = (_Float16*)alloc(sizeof(_Float16) * (size_t)BLD);
    _Float16* WinT  = (_Float16*)alloc(sizeof(_Float16) * (size_t)C3 * Dc);
    _Float16* v2T   = (_Float16*)alloc(sizeof(_Float16) * (size_t)BLD);
    _Float16* WoutT = (_Float16*)alloc(sizeof(_Float16) * (size_t)Dc * Dc);

    // 0) f16 operand preparation (halves GEMM HBM traffic, enables TDM staging)
    convert_f16<<<dim3((unsigned)(BLD / 256)), dim3(256), 0, stream>>>(u, uh, BLD);
    transpose_convert_f16<<<dim3(C3 / 32, Dc / 32, 1), dim3(256), 0, stream>>>(
        W_in, 0, WinT, 0, Dc, C3);
    transpose_convert_f16<<<dim3(Dc / 32, Dc / 32, 1), dim3(256), 0, stream>>>(
        W_out, 0, WoutT, 0, Dc, Dc);

    // 1) truncated Hyena filter taps
    build_filter<<<dim3(KF), dim3(256), 0, stream>>>(W1, b1, W2, b2, f_decay, g);

    // 2) proj = u @ W_in + b_in   (WMMA)
    gemm_wmma_f16<<<dim3(C3 / BN, (Bc * Lc) / BM, 1), dim3(256), 0, stream>>>(
        uh, 0, WinT, b_in, proj, 0, Bc * Lc, C3, Dc);

    // 3) conv3 + transpose -> xv (B,3,D,L)
    conv3_transpose<<<dim3(C3 / 32, Lc / 32, Bc), dim3(256), 0, stream>>>(
        proj, conv_k, conv_b, xv);

    // 4) order 0: S = cumsum(v); v1 = (f_bias0*S + g0*v) * x0  -> yb
    row_scan<<<dim3(Bc * Dc), dim3(256), 0, stream>>>(xv + 2 * DL, XV3, Sb);
    fir_combine<<<dim3(Lc / 256, Dc, Bc), dim3(256), 0, stream>>>(
        xv + 2 * DL, XV3, Sb, xv, XV3, g, f_bias, yb, DL);

    // 5) order 1: S = cumsum(v1); v2 = (f_bias1*S + g1*v1) * x1 -> xv[:,2]
    row_scan<<<dim3(Bc * Dc), dim3(256), 0, stream>>>(yb, DL, Sb);
    fir_combine<<<dim3(Lc / 256, Dc, Bc), dim3(256), 0, stream>>>(
        yb, DL, Sb, xv + 1 * DL, XV3, g + (long)Dc * KF, f_bias + Dc,
        xv + 2 * DL, XV3);

    // 6) v2 (B,D,L) -> v2T (B,L,D) f16, then out = v2T @ W_out + b_out (WMMA)
    transpose_convert_f16<<<dim3(Lc / 32, Dc / 32, Bc), dim3(256), 0, stream>>>(
        xv + 2 * DL, XV3, v2T, DL, Dc, Lc);
    gemm_wmma_f16<<<dim3(Dc / BN, Lc / BM, Bc), dim3(256), 0, stream>>>(
        v2T, DL, WoutT, b_out, out, DL, Lc, Dc, Dc);
}
